// CCE_Model_77936476553266
// MI455X (gfx1250) — compile-verified, matrix operand order checked by reading, not run on hardware
//
#include <hip/hip_runtime.h>
#include <hip/hip_bf16.h>
#include <math.h>

typedef _Float16 f16;
typedef __attribute__((ext_vector_type(16))) _Float16 v16h;
typedef __attribute__((ext_vector_type(8)))  _Float16 v8h;
typedef __attribute__((ext_vector_type(8)))  float    v8f;

union AFrag { v16h v; struct { v8h lo, hi; } s; };

#define BN_EPS 1e-5f

// ---------------------------------------------------------------------------
// Packing / elementwise kernels
// ---------------------------------------------------------------------------

// OIHW f32 -> [O, Kpad] f16 with k = (kh*KW+kw)*C + c  (works for 1x1 too)
__global__ void k_repack_conv_w(const float* __restrict__ w, f16* __restrict__ wt,
                                int O, int C, int KH, int KW, int Kpad)
{
    long long i = (long long)blockIdx.x * blockDim.x + threadIdx.x;
    long long tot = (long long)O * Kpad;
    if (i >= tot) return;
    int k = (int)(i % Kpad);
    int o = (int)(i / Kpad);
    int K = C * KH * KW;
    float val = 0.f;
    if (k < K) {
        int c = k % C; int t = k / C;
        int kw = t % KW; int kh = t / KW;
        val = w[(((long long)o * C + c) * KH + kh) * KW + kw];
    }
    wt[i] = (f16)val;
}

// fc_w [512,8192] with k_src = c*16 + hw  ->  wt [512,8192] with k_dst = hw*512 + c
__global__ void k_repack_fc_w(const float* __restrict__ w, f16* __restrict__ wt)
{
    long long i = (long long)blockIdx.x * blockDim.x + threadIdx.x;
    if (i >= (long long)512 * 8192) return;
    int k = (int)(i % 8192); int n = (int)(i / 8192);
    int hw = k / 512; int c = k % 512;
    wt[i] = (f16)w[(long long)n * 8192 + c * 16 + hw];
}

__global__ void k_bn_fold(const float* __restrict__ g, const float* __restrict__ b,
                          const float* __restrict__ m, const float* __restrict__ v,
                          float* __restrict__ scale, float* __restrict__ bias, int C)
{
    int c = blockIdx.x * blockDim.x + threadIdx.x;
    if (c >= C) return;
    float s = g[c] * rsqrtf(v[c] + BN_EPS);
    scale[c] = s;
    bias[c]  = b[c] - m[c] * s;
}

// x [B,3,32,32] f32 NCHW -> col [B*16*16, 160] f16; K=147, k=(kh*7+kw)*3+c
__global__ void k_im2col_conv1(const float* __restrict__ x, f16* __restrict__ col, int B)
{
    long long i = (long long)blockIdx.x * blockDim.x + threadIdx.x;
    long long tot = (long long)B * 256 * 160;
    if (i >= tot) return;
    int k = (int)(i % 160);
    long long mrow = i / 160;
    float val = 0.f;
    if (k < 147) {
        int c = k % 3; int t = k / 3;
        int kw = t % 7; int kh = t / 7;
        int hw = (int)(mrow % 256); int n = (int)(mrow / 256);
        int oh = hw >> 4, ow = hw & 15;
        int ih = oh * 2 - 3 + kh, iw = ow * 2 - 3 + kw;
        if (ih >= 0 && ih < 32 && iw >= 0 && iw < 32)
            val = x[(((long long)n * 3 + c) * 32 + ih) * 32 + iw];
    }
    col[i] = (f16)val;
}

// in [B,H,W,C] f16 -> col [B*OH*OW, 9*C], pad=1, 3x3, stride s; k=(kh*3+kw)*C+c
__global__ void k_im2col_3x3(const f16* __restrict__ in, f16* __restrict__ col,
                             int B, int H, int W, int C, int stride, int OH, int OW)
{
    long long i = (long long)blockIdx.x * blockDim.x + threadIdx.x;
    long long tot = (long long)B * OH * OW * 9 * C;
    if (i >= tot) return;
    int c = (int)(i % C); long long t = i / C;
    int kw = (int)(t % 3); t /= 3;
    int kh = (int)(t % 3); t /= 3;
    int ow = (int)(t % OW); t /= OW;
    int oh = (int)(t % OH); int n = (int)(t / OH);
    int ih = oh * stride - 1 + kh, iw = ow * stride - 1 + kw;
    f16 val = (f16)0.f;
    if (ih >= 0 && ih < H && iw >= 0 && iw < W)
        val = in[(((long long)n * H + ih) * W + iw) * C + c];
    long long mrow = ((long long)n * OH + oh) * OW + ow;
    col[mrow * (9 * C) + (kh * 3 + kw) * C + c] = val;
}

// 3x3 stride-2 pad-1 maxpool on [B,H,H,C] f16 -> [B,H/2,H/2,C]
__global__ void k_maxpool(const f16* __restrict__ in, f16* __restrict__ out,
                          int B, int H, int C)
{
    int OH = H / 2;
    long long i = (long long)blockIdx.x * blockDim.x + threadIdx.x;
    long long tot = (long long)B * OH * OH * C;
    if (i >= tot) return;
    int c = (int)(i % C); long long t = i / C;
    int ow = (int)(t % OH); t /= OH;
    int oh = (int)(t % OH); int n = (int)(t / OH);
    float best = -3.0e38f;
    for (int kh = 0; kh < 3; ++kh)
        for (int kw = 0; kw < 3; ++kw) {
            int ih = oh * 2 - 1 + kh, iw = ow * 2 - 1 + kw;
            if (ih >= 0 && ih < H && iw >= 0 && iw < H) {
                float v = (float)in[(((long long)n * H + ih) * H + iw) * C + c];
                best = fmaxf(best, v);
            }
        }
    out[i] = (f16)best;
}

// stride-2 spatial subsample of [B,H,H,C] -> [B,H/2,H/2,C]
__global__ void k_subsample2(const f16* __restrict__ in, f16* __restrict__ out,
                             int B, int H, int C)
{
    int OH = H / 2;
    long long i = (long long)blockIdx.x * blockDim.x + threadIdx.x;
    long long tot = (long long)B * OH * OH * C;
    if (i >= tot) return;
    int c = (int)(i % C); long long t = i / C;
    int ow = (int)(t % OH); t /= OH;
    int oh = (int)(t % OH); int n = (int)(t / OH);
    out[i] = in[(((long long)n * H + 2 * oh) * H + 2 * ow) * C + c];
}

__global__ void k_zero(float* p, int n)
{
    int i = blockIdx.x * blockDim.x + threadIdx.x;
    if (i < n) p[i] = 0.f;
}

// ---------------------------------------------------------------------------
// Fused WMMA GEMM with async-LDS staged B operand.
//   out[m,n] = act( sum_k A[m,k]*Wt[n,k] * scale[n] + bias[n] + resid[m,n] )
// Block = 128 threads = 4 waves; block computes 64(M) x 64(N); each wave one
// 16x64 strip. B chunk (64 rows x 32 halves) double-buffered in LDS via
// global_load_async_to_lds_b128; all 4 waves share it.
// Requires M%64==0, N%64==0, K%32==0.
// ---------------------------------------------------------------------------
#define LDSROWB 80                       // bytes per LDS B row (64B data + pad)
#define LDSBUFB (64 * LDSROWB)           // 5120 B per buffer

__global__ void __launch_bounds__(128)
k_gemm_wmma(const f16* __restrict__ A, const f16* __restrict__ Wt,
            int M, int N, int K,
            const float* __restrict__ scale, const float* __restrict__ bias,
            const f16* __restrict__ resid,
            f16* __restrict__ outh, float* __restrict__ outf, int relu)
{
    __shared__ f16 lds[2 * LDSBUFB / 2];

    int tid  = threadIdx.x;
    int lane = tid & 31;
    int wv   = tid >> 5;
    int nl   = lane & 15;
    int koff = (lane >> 4) * 8;          // halves offset for this lane group
    int n0   = blockIdx.x * 64;
    int tm   = blockIdx.y * 4 + wv;

    const f16* arow = A + (long long)(tm * 16 + nl) * K + koff;

    // Each thread stages two 16B chunks of the 64x64B B tile per K-step.
    int ch0 = tid, ch1 = tid + 128;      // chunk = row*4 + seg
    int row0 = ch0 >> 2, seg0 = ch0 & 3;
    int row1 = ch1 >> 2, seg1 = ch1 & 3;

    // LDS aperture: low 32 bits of a generic pointer into LDS are the
    // hardware LDS byte address (ISA 10.2).
    // Deriving the asm operand from the real array pointer also captures the
    // array (ptrtoint), so the "memory"-clobbering asm is a potential writer
    // and the ds_load reads below cannot be folded away.
    char* ldsb = (char*)lds;
    auto stage = [&](int buf, int k0) {
        unsigned l0 = (unsigned)(uintptr_t)(ldsb + buf * LDSBUFB + row0 * LDSROWB + seg0 * 16);
        unsigned l1 = (unsigned)(uintptr_t)(ldsb + buf * LDSBUFB + row1 * LDSROWB + seg1 * 16);
        const f16* g0 = Wt + (long long)(n0 + row0) * K + k0 + seg0 * 8;
        const f16* g1 = Wt + (long long)(n0 + row1) * K + k0 + seg1 * 8;
        asm volatile("global_load_async_to_lds_b128 %0, %1, off"
                     :: "v"(l0), "v"(g0) : "memory");
        asm volatile("global_load_async_to_lds_b128 %0, %1, off"
                     :: "v"(l1), "v"(g1) : "memory");
    };

    stage(0, 0);
    asm volatile("s_wait_asynccnt 0x0" ::: "memory");
    __syncthreads();

    v8f acc[4] = {};
    int nk = K >> 5;
    for (int i = 0; i < nk; ++i) {
        int cur = i & 1;
        if (i + 1 < nk) stage(cur ^ 1, (i + 1) << 5);

        AFrag a;
        a.s.lo = *(const v8h*)(arow + (i << 5));
        a.s.hi = *(const v8h*)(arow + (i << 5) + 16);

        const char* base = ldsb + cur * LDSBUFB;
#pragma unroll
        for (int t = 0; t < 4; ++t) {
            const char* bp = base + (t * 16 + nl) * LDSROWB + koff * 2;
            AFrag b;
            b.s.lo = *(const v8h*)(bp);
            b.s.hi = *(const v8h*)(bp + 32);
            acc[t] = __builtin_amdgcn_wmma_f32_16x16x32_f16(
                false, a.v, false, b.v, (short)0, acc[t], false, false);
        }
        asm volatile("s_wait_asynccnt 0x0" ::: "memory");
        __syncthreads();
    }

    int mbase = tm * 16 + (lane >> 4) * 8;   // rows m = mbase..mbase+7 <-> vgpr 0..7
#pragma unroll
    for (int t = 0; t < 4; ++t) {
        int n = n0 + t * 16 + nl;
        float sc = scale ? scale[n] : 1.f;
        float bs = bias  ? bias[n]  : 0.f;
#pragma unroll
        for (int r = 0; r < 8; ++r) {
            long long idx = (long long)(mbase + r) * N + n;
            float val = acc[t][r] * sc + bs;
            if (resid) val += (float)resid[idx];
            if (relu)  val = fmaxf(val, 0.f);
            if (outh)  outh[idx] = (f16)val;
            if (outf)  outf[idx] = val;
        }
    }
}

// ---------------------------------------------------------------------------
// Head: stereographic projection, fc2 logits, log-softmax loss over proj(257)
// ---------------------------------------------------------------------------
__global__ void k_head(const float* __restrict__ f, const float* __restrict__ fc2w,
                       const int* __restrict__ labels, float* __restrict__ out,
                       float* __restrict__ loss, int B)
{
    int r = blockIdx.x * blockDim.x + threadIdx.x;
    if (r >= B) return;
    const float* fr = f + (long long)r * 256;
    float sq = 0.f;
    for (int i = 0; i < 256; ++i) sq += fr[i] * fr[i];
    float s = -(1.f - sq) / (1.f + sq);
    float a = 1.f - s;                 // proj[k<256] = a*fr[k], proj[256] = s
    for (int c = 0; c < 10; ++c) {
        const float* wr = fc2w + c * 257;
        float acc = wr[256] * s;
        for (int k = 0; k < 256; ++k) acc += wr[k] * (a * fr[k]);
        out[r * 10 + c] = acc;
    }
    float mx = s;
    for (int k = 0; k < 256; ++k) mx = fmaxf(mx, a * fr[k]);
    float se = 0.f;
    for (int k = 0; k < 256; ++k) se += expf(a * fr[k] - mx);
    se += expf(s - mx);
    int lb = labels[r];
    float pl = (lb < 256) ? a * fr[lb] : s;
    float lp = pl - mx - logf(se);
    atomicAdd(loss, -lp / (float)B);
}

// ---------------------------------------------------------------------------
// Host orchestration
// ---------------------------------------------------------------------------
#define LAUNCH1D(kern, total, ...)                                              \
    do {                                                                        \
        long long _t = (long long)(total);                                      \
        unsigned _g = (unsigned)((_t + 255) / 256);                             \
        kern<<<dim3(_g), dim3(256), 0, stream>>>(__VA_ARGS__);                  \
    } while (0)

struct BNp { const float *g, *b, *m, *v; };

extern "C" void kernel_launch(void* const* d_in, const int* in_sizes, int n_in,
                              void* d_out, int out_size, void* d_ws, size_t ws_size,
                              hipStream_t stream)
{
    (void)in_sizes; (void)n_in; (void)out_size; (void)ws_size;
    const int B = 2048;

    // -------- walk inputs in setup_inputs() dict order --------
    int pi = 0;
    auto F = [&]() { return (const float*)d_in[pi++]; };
    const float* X = F();                       // x [2048,3,32,32]
    const float* conv1w = F();
    BNp bn1 = {F(), F(), F(), F()};

    struct Blk { const float *w1, *w2, *w3, *wd; BNp b1, b2, b3, bd; };
    Blk blk[7];
    for (int i = 0; i < 7; ++i) {
        bool down = (i == 0 || i == 3);
        Blk& bb = blk[i];
        bb.w1 = F(); bb.b1 = {F(), F(), F(), F()};
        bb.w2 = F(); bb.b2 = {F(), F(), F(), F()};
        bb.w3 = F(); bb.b3 = {F(), F(), F(), F()};
        if (down) { bb.wd = F(); bb.bd = {F(), F(), F(), F()}; }
        else       { bb.wd = nullptr; bb.bd = {nullptr, nullptr, nullptr, nullptr}; }
    }
    const float* fc_w  = F();
    const float* fc_b  = F();
    const float* fc1_w = F();
    const float* fc1_b = F();
    const float* fc2_w = F();
    const int* labels = (const int*)d_in[pi++];

    // -------- workspace bump allocator --------
    char* wsb = (char*)d_ws;
    size_t off = 0;
    auto alloc = [&](size_t bytes) -> void* {
        void* p = wsb + off;
        off += (bytes + 255) & ~(size_t)255;
        return p;
    };

    struct Cfg { int cin, width, Hin, stride, down; };
    const Cfg cfg[7] = {
        {64, 64, 8, 1, 1}, {256, 64, 8, 1, 0}, {256, 64, 8, 1, 0},
        {256, 128, 8, 2, 1}, {512, 128, 4, 1, 0}, {512, 128, 4, 1, 0}, {512, 128, 4, 1, 0}
    };

    // packed weights + folded BN
    f16* Wc1 = (f16*)alloc((size_t)64 * 160 * 2);
    float* sC1 = (float*)alloc(64 * 4);
    float* bC1 = (float*)alloc(64 * 4);
    f16 *pW1[7], *pW2[7], *pW3[7], *pWd[7];
    float *sc1[7], *bi1[7], *sc2[7], *bi2[7], *sc3[7], *bi3[7], *scd[7], *bid[7];
    for (int i = 0; i < 7; ++i) {
        int w = cfg[i].width, cin = cfg[i].cin, O = 4 * w;
        pW1[i] = (f16*)alloc((size_t)w * cin * 2);
        pW2[i] = (f16*)alloc((size_t)w * 9 * w * 2);
        pW3[i] = (f16*)alloc((size_t)O * w * 2);
        sc1[i] = (float*)alloc(w * 4);  bi1[i] = (float*)alloc(w * 4);
        sc2[i] = (float*)alloc(w * 4);  bi2[i] = (float*)alloc(w * 4);
        sc3[i] = (float*)alloc(O * 4);  bi3[i] = (float*)alloc(O * 4);
        if (cfg[i].down) {
            pWd[i] = (f16*)alloc((size_t)O * cin * 2);
            scd[i] = (float*)alloc(O * 4);  bid[i] = (float*)alloc(O * 4);
        } else { pWd[i] = nullptr; scd[i] = bid[i] = nullptr; }
    }
    f16* Wfc  = (f16*)alloc((size_t)512 * 8192 * 2);
    f16* Wfc1 = (f16*)alloc((size_t)256 * 512 * 2);

    // big buffers
    const long long MC1 = (long long)B * 256;          // 524288 rows after conv1
    f16* COL = (f16*)alloc((size_t)MC1 * 160 * 2);     // max im2col (conv1)
    size_t ACTB = (size_t)MC1 * 64 * 2;                // 64 MiB act slot
    f16* S[5];
    for (int i = 0; i < 5; ++i) S[i] = (f16*)alloc(ACTB);
    float* F1 = (float*)alloc((size_t)B * 256 * 4);    // fc1 output (f32)

    // -------- weight repack + BN fold --------
    LAUNCH1D(k_repack_conv_w, (long long)64 * 160, conv1w, Wc1, 64, 3, 7, 7, 160);
    LAUNCH1D(k_bn_fold, 64, bn1.g, bn1.b, bn1.m, bn1.v, sC1, bC1, 64);
    for (int i = 0; i < 7; ++i) {
        int w = cfg[i].width, cin = cfg[i].cin, O = 4 * w;
        LAUNCH1D(k_repack_conv_w, (long long)w * cin,   blk[i].w1, pW1[i], w, cin, 1, 1, cin);
        LAUNCH1D(k_repack_conv_w, (long long)w * 9 * w, blk[i].w2, pW2[i], w, w, 3, 3, 9 * w);
        LAUNCH1D(k_repack_conv_w, (long long)O * w,     blk[i].w3, pW3[i], O, w, 1, 1, w);
        LAUNCH1D(k_bn_fold, w, blk[i].b1.g, blk[i].b1.b, blk[i].b1.m, blk[i].b1.v, sc1[i], bi1[i], w);
        LAUNCH1D(k_bn_fold, w, blk[i].b2.g, blk[i].b2.b, blk[i].b2.m, blk[i].b2.v, sc2[i], bi2[i], w);
        LAUNCH1D(k_bn_fold, O, blk[i].b3.g, blk[i].b3.b, blk[i].b3.m, blk[i].b3.v, sc3[i], bi3[i], O);
        if (cfg[i].down) {
            LAUNCH1D(k_repack_conv_w, (long long)O * cin, blk[i].wd, pWd[i], O, cin, 1, 1, cin);
            LAUNCH1D(k_bn_fold, O, blk[i].bd.g, blk[i].bd.b, blk[i].bd.m, blk[i].bd.v, scd[i], bid[i], O);
        }
    }
    LAUNCH1D(k_repack_fc_w, (long long)512 * 8192, fc_w, Wfc);
    LAUNCH1D(k_repack_conv_w, (long long)256 * 512, fc1_w, Wfc1, 256, 512, 1, 1, 512);

    auto gemm = [&](const f16* A, const f16* Wt, int M, int N, int K,
                    const float* sc, const float* bs, const f16* res,
                    f16* oh, float* of, int relu) {
        dim3 g((unsigned)(N / 64), (unsigned)(M / 64)), b(128);
        k_gemm_wmma<<<g, b, 0, stream>>>(A, Wt, M, N, K, sc, bs, res, oh, of, relu);
    };

    // -------- conv1 + maxpool --------
    LAUNCH1D(k_im2col_conv1, MC1 * 160, X, COL, B);
    gemm(COL, Wc1, (int)MC1, 64, 160, sC1, bC1, nullptr, S[0], nullptr, 1);
    LAUNCH1D(k_maxpool, (long long)B * 64 * 64, S[0], S[1], B, 16, 64);

    f16* cur = S[1];
    int curIdx = 1;

    // -------- bottleneck blocks --------
    for (int i = 0; i < 7; ++i) {
        const Cfg c = cfg[i];
        int w = c.width, O = 4 * w;
        int Hout = c.Hin / c.stride;
        int Min  = B * c.Hin * c.Hin;
        int Mout = B * Hout * Hout;

        int fr[4], nf = 0;
        for (int s = 0; s < 5; ++s) if (s != curIdx) fr[nf++] = s;
        f16* t1   = S[fr[0]];
        f16* t2   = S[fr[1]];
        f16* idb  = S[fr[2]];
        f16* outb = S[fr[3]];

        const f16* resid;
        if (c.down) {
            const f16* idsrc = cur;
            if (c.stride == 2) {
                LAUNCH1D(k_subsample2, (long long)Mout * c.cin, cur, t2, B, c.Hin, c.cin);
                idsrc = t2;
            }
            gemm(idsrc, pWd[i], Mout, O, c.cin, scd[i], bid[i], nullptr, idb, nullptr, 0);
            resid = idb;
        } else {
            resid = cur;
        }
        gemm(cur, pW1[i], Min, w, c.cin, sc1[i], bi1[i], nullptr, t1, nullptr, 1);
        LAUNCH1D(k_im2col_3x3, (long long)Mout * 9 * w, t1, COL, B, c.Hin, c.Hin, w,
                 c.stride, Hout, Hout);
        gemm(COL, pW2[i], Mout, w, 9 * w, sc2[i], bi2[i], nullptr, t2, nullptr, 1);
        gemm(t2, pW3[i], Mout, O, w, sc3[i], bi3[i], resid, outb, nullptr, 1);

        cur = outb;
        curIdx = fr[3];
    }

    // -------- FC head --------
    // feat [B*16, 512] contiguous == [B, 8192] with k = hw*512 + c (fc_w repacked to match)
    f16* fco = S[(curIdx + 1) % 5];
    gemm(cur, Wfc, B, 512, 8192, nullptr, fc_b, nullptr, fco, nullptr, 1);
    gemm(fco, Wfc1, B, 256, 512, nullptr, fc1_b, nullptr, nullptr, F1, 1);

    float* out = (float*)d_out;
    k_zero<<<1, 32, 0, stream>>>(out + B * 10, 1);
    k_head<<<dim3((B + 255) / 256), dim3(256), 0, stream>>>(F1, fc2_w, labels, out,
                                                            out + B * 10, B);
}
